// ARCUnstacked_12558484374031
// MI455X (gfx1250) — compile-verified, matrix operand order checked by reading, not run on hardware
//
#include <hip/hip_runtime.h>
#include <stdint.h>

#define BATCH 4
#define TT    4096      // STATE_LEN + SEQ_LEN + STATE_LEN
#define DIN   512
#define DHD   64
#define QSCALE 0.125f

typedef __attribute__((ext_vector_type(2)))  float  v2f;
typedef __attribute__((ext_vector_type(8)))  float  v8f;
typedef __attribute__((ext_vector_type(8)))  __bf16 v8bf;
typedef __attribute__((ext_vector_type(16))) __bf16 v16bf;

// Pack two contiguous 8x-bf16 (16B) chunks into one 16-element WMMA fragment.
// For 16-bit A/B fragments: elements 0..7 <-> K = 8h+0..7, elements 8..15 <-> K = 16+8h+0..7.
__device__ __forceinline__ v16bf pack16(const __bf16* base, int off_lo, int off_hi) {
  v8bf lo = *reinterpret_cast<const v8bf*>(base + off_lo);
  v8bf hi = *reinterpret_cast<const v8bf*>(base + off_hi);
  v16bf r;
#pragma unroll
  for (int i = 0; i < 8; ++i) { r[i] = lo[i]; r[8 + i] = hi[i]; }
  return r;
}

// CDNA5 async global->LDS copy (16B per lane), tracked by ASYNCcnt.
__device__ __forceinline__ void async_g2l_b128(uint32_t lds_off, const void* gptr) {
  asm volatile("global_load_async_to_lds_b128 %0, %1, off"
               :: "v"(lds_off), "v"(gptr) : "memory");
}
__device__ __forceinline__ void wait_async0() {
  asm volatile("s_wait_asynccnt 0x0" ::: "memory");
}

// ---------------------------------------------------------------------------
// Kernel 1: fused QKV projection with fp32 WMMA (16x16x4).
// One wave computes a 16(token) x 16(col) tile for Q, K and V simultaneously
// (x A-fragments loaded ONCE -> 1x HBM read of the 32MB x tensor instead of 3x).
// Q is written scaled by QSCALE; V is written transposed ([d][token]).
// ---------------------------------------------------------------------------
__global__ __launch_bounds__(128) void proj_kernel(
    const float* __restrict__ x,
    const float* __restrict__ Wq, const float* __restrict__ Wk, const float* __restrict__ Wv,
    __bf16* __restrict__ Qb, __bf16* __restrict__ Kb, __bf16* __restrict__ Vt)
{
  const int lane = threadIdx.x & 31;
  const int wv   = threadIdx.x >> 5;     // n-tile 0..3
  const int tt   = blockIdx.x % (TT / 16);
  const int b    = blockIdx.x / (TT / 16);
  const int t0   = tt * 16, n0 = wv * 16;
  const int h    = lane >> 4, ml = lane & 15;

  const float* xrow = x + ((size_t)(b * TT + t0 + ml)) * DIN;
  v8f cq = {0.f,0.f,0.f,0.f,0.f,0.f,0.f,0.f};
  v8f ck = {0.f,0.f,0.f,0.f,0.f,0.f,0.f,0.f};
  v8f cv = {0.f,0.f,0.f,0.f,0.f,0.f,0.f,0.f};
  for (int k0 = 0; k0 < DIN; k0 += 4) {
    const int kk = k0 + 2 * h;           // f32 A 16x4: lanes 0-15 hold K0,K1; lanes 16-31 K2,K3
    v2f a;  a.x = xrow[kk];  a.y = xrow[kk + 1];
    const size_t w0 = (size_t)kk * DHD + n0 + ml;
    v2f bq; bq.x = Wq[w0]; bq.y = Wq[w0 + DHD];
    cq = __builtin_amdgcn_wmma_f32_16x16x4_f32(false, a, false, bq, (short)0, cq, false, false);
    v2f bk; bk.x = Wk[w0]; bk.y = Wk[w0 + DHD];
    ck = __builtin_amdgcn_wmma_f32_16x16x4_f32(false, a, false, bk, (short)0, ck, false, false);
    v2f bv; bv.x = Wv[w0]; bv.y = Wv[w0 + DHD];
    cv = __builtin_amdgcn_wmma_f32_16x16x4_f32(false, a, false, bv, (short)0, cv, false, false);
  }

  // Q, K: row-major bf16 stores (C layout: M = reg + 8*(lane>=16), N = lane&15)
#pragma unroll
  for (int r = 0; r < 8; ++r) {
    const int m = r + 8 * h;
    const size_t rowoff = ((size_t)(b * TT + t0 + m)) * DHD + n0 + ml;
    Qb[rowoff] = (__bf16)(cq[r] * QSCALE);
    Kb[rowoff] = (__bf16)ck[r];
  }
  // V: transposed store; registers r=0..7 are 8 consecutive tokens -> one b128 store.
  v8bf pv;
#pragma unroll
  for (int r = 0; r < 8; ++r) pv[r] = (__bf16)cv[r];
  *reinterpret_cast<v8bf*>(Vt + ((size_t)(b * DHD + n0 + ml)) * TT + t0 + 8 * h) = pv;
}

// ---------------------------------------------------------------------------
// Kernel 1b: recompute the 2*STATE_LEN boundary rows with the *_s weights.
// ---------------------------------------------------------------------------
__global__ __launch_bounds__(64) void fixup_kernel(
    const float* __restrict__ x,
    const float* __restrict__ Wq_s, const float* __restrict__ Wk_s, const float* __restrict__ Wv_s,
    __bf16* __restrict__ Qb, __bf16* __restrict__ Kb, __bf16* __restrict__ Vt)
{
  const int n   = threadIdx.x;                       // output column
  const int rid = blockIdx.x & 15;
  const int b   = blockIdx.x >> 4;
  const int row = (rid < 8) ? rid : (4080 + rid);    // rows 0..7 and 4088..4095
  const float* xr = x + ((size_t)(b * TT + row)) * DIN;
  float aq = 0.f, ak = 0.f, av = 0.f;
  for (int i = 0; i < DIN; ++i) {
    const float xv = xr[i];
    aq += xv * Wq_s[(size_t)i * DHD + n];
    ak += xv * Wk_s[(size_t)i * DHD + n];
    av += xv * Wv_s[(size_t)i * DHD + n];
  }
  Qb[((size_t)(b * TT + row)) * DHD + n] = (__bf16)(aq * QSCALE);
  Kb[((size_t)(b * TT + row)) * DHD + n] = (__bf16)ak;
  Vt[((size_t)(b * DHD + n)) * TT + row] = (__bf16)av;
}

// ---------------------------------------------------------------------------
// Kernel 2: causal flash attention, bf16 WMMA 16x16x32, fp32 accumulate.
// K/V tiles staged in LDS with double-buffered async copies (ASYNCcnt):
// the block loads each 8KB K/V tile ONCE per 32-key step instead of 4x.
// Wave handles 32 queries (two 16-wide N-tiles); S^T = K.Q^T so exp(S)
// repacks into the P B-fragment with zero cross-lane movement.
// ---------------------------------------------------------------------------
__global__ __launch_bounds__(128) void attn_kernel(
    const __bf16* __restrict__ Qb, const __bf16* __restrict__ Kb,
    const __bf16* __restrict__ Vt, float* __restrict__ out)
{
  // Padded LDS tiles: row strides 72/40 elements -> 16-lane b128 reads are conflict-free.
  __shared__ __bf16 Ksh[2][32][72];   // [buf][key][dim]
  __shared__ __bf16 Vsh[2][64][40];   // [buf][dim][key]

  const int tid  = threadIdx.x;
  const int lane = tid & 31;
  const int wv   = tid >> 5;
  const int b    = blockIdx.y;
  // reversed mapping: longest (highest-query) causal blocks launch first
  const int mblk = (gridDim.x - 1 - blockIdx.x) * 128;   // block's 128-query base
  const int m0   = mblk + wv * 32;                       // wave's 32-query base
  const int h    = lane >> 4, ql = lane & 15;

  const size_t kgbase = (size_t)(b * TT) * DHD;          // Kb row-major base
  const size_t vgbase = (size_t)(b * DHD) * TT;          // Vt transposed base

  // Cooperative async tile loaders: 128 threads x 2 passes x 16B = 4KB each.
  auto loadK = [&](int buf, int kt0) {
#pragma unroll
    for (int p = 0; p < 2; ++p) {
      const int i  = p * 128 + tid;          // 0..255
      const int kr = i >> 3, ch = (i & 7) * 8;
      async_g2l_b128((uint32_t)(uintptr_t)&Ksh[buf][kr][ch],
                     Kb + kgbase + (size_t)(kt0 + kr) * DHD + ch);
    }
  };
  auto loadV = [&](int buf, int kt0) {
#pragma unroll
    for (int p = 0; p < 2; ++p) {
      const int i = p * 128 + tid;           // 0..255
      const int d = i >> 2, ch = (i & 3) * 8;
      async_g2l_b128((uint32_t)(uintptr_t)&Vsh[buf][d][ch],
                     Vt + vgbase + (size_t)d * TT + kt0 + ch);
    }
  };

  // Q as B-fragments of S^T = K.Q^T: [qt][dim-block]
  v16bf qf[2][2];
#pragma unroll
  for (int qt = 0; qt < 2; ++qt) {
    const __bf16* qrow = Qb + ((size_t)(b * TT + m0 + 16 * qt + ql)) * DHD;
    qf[qt][0] = pack16(qrow, 8 * h,      16 + 8 * h);
    qf[qt][1] = pack16(qrow, 32 + 8 * h, 48 + 8 * h);
  }

  v8f o[2][4];
#pragma unroll
  for (int qt = 0; qt < 2; ++qt)
#pragma unroll
    for (int t = 0; t < 4; ++t) o[qt][t] = (v8f){0.f,0.f,0.f,0.f,0.f,0.f,0.f,0.f};
  float mq[2] = {-3.0e38f, -3.0e38f};
  float lq[2] = {0.0f, 0.0f};

  // Lockstep trip count for the whole block (covers keys 0 .. mblk+127).
  const int nbl = (mblk + 159) >> 5;
  loadK(0, 0);
  loadV(0, 0);

  for (int jb = 0; jb < nbl; ++jb) {
    const int cur = jb & 1;
    const int kt0 = jb << 5;

    wait_async0();            // my async copies into buf[cur] are done
    __syncthreads();          // everyone's copies done; everyone done reading buf[cur^1]

    if (jb + 1 < nbl) {       // start filling the other buffer
      loadK(cur ^ 1, (jb + 1) << 5);
      loadV(cur ^ 1, (jb + 1) << 5);
    }

    if (kt0 <= m0 + 31) {     // wave-uniform: skip fully-masked trailing blocks
      // S^T tiles: keys along M (registers), queries along N (lanes); s[qt][u]
      v8f s[2][2];
#pragma unroll
      for (int u = 0; u < 2; ++u) {
        const __bf16* krow = &Ksh[cur][16 * u + ql][0];
        const v16bf kf0 = pack16(krow, 8 * h,      16 + 8 * h);
        const v16bf kf1 = pack16(krow, 32 + 8 * h, 48 + 8 * h);
#pragma unroll
        for (int qt = 0; qt < 2; ++qt) {
          v8f acc = {0.f,0.f,0.f,0.f,0.f,0.f,0.f,0.f};
          acc = __builtin_amdgcn_wmma_f32_16x16x32_bf16(false, kf0, false, qf[qt][0], (short)0, acc, false, false);
          acc = __builtin_amdgcn_wmma_f32_16x16x32_bf16(false, kf1, false, qf[qt][1], (short)0, acc, false, false);
          s[qt][u] = acc;
        }
      }

      // Per query-tile: causal mask + online softmax + P fragment
      v16bf pB[2];
#pragma unroll
      for (int qt = 0; qt < 2; ++qt) {
        const int qbase = m0 + 16 * qt;
        const int q     = qbase + ql;
        if (kt0 + 31 > qbase) {                // only diagonal-straddling blocks
#pragma unroll
          for (int u = 0; u < 2; ++u)
#pragma unroll
            for (int r = 0; r < 8; ++r) {
              const int kj = kt0 + 16 * u + r + 8 * h;
              if (kj > q) s[qt][u][r] = -1.0e30f;
            }
        }
        float rm = -3.0e38f;
#pragma unroll
        for (int u = 0; u < 2; ++u)
#pragma unroll
          for (int r = 0; r < 8; ++r) rm = fmaxf(rm, s[qt][u][r]);
        rm = fmaxf(rm, __shfl_xor(rm, 16, 32)); // stats replicated in lanes l, l^16
        const float mn    = fmaxf(mq[qt], rm);
        const float alpha = __expf(mq[qt] - mn);
        mq[qt] = mn;

        float rs = 0.0f;
#pragma unroll
        for (int u = 0; u < 2; ++u)
#pragma unroll
          for (int r = 0; r < 8; ++r) {
            const float p = __expf(s[qt][u][r] - mn);
            rs += p;
            pB[qt][u * 8 + r] = (__bf16)p;      // element e <-> key kt0+16u+8h+e
          }
        rs += __shfl_xor(rs, 16, 32);
        lq[qt] = lq[qt] * alpha + rs;

#pragma unroll
        for (int t = 0; t < 4; ++t)
#pragma unroll
          for (int r = 0; r < 8; ++r) o[qt][t][r] *= alpha;
      }

      // O^T += V^T . P : each V^T A-fragment (from LDS) feeds both query tiles
#pragma unroll
      for (int t = 0; t < 4; ++t) {
        const __bf16* vrow = &Vsh[cur][16 * t + ql][0];
        const v16bf vf = pack16(vrow, 8 * h, 16 + 8 * h);
        o[0][t] = __builtin_amdgcn_wmma_f32_16x16x32_bf16(false, vf, false, pB[0], (short)0, o[0][t], false, false);
        o[1][t] = __builtin_amdgcn_wmma_f32_16x16x32_bf16(false, vf, false, pB[1], (short)0, o[1][t], false, false);
      }
    }
  }

  // Epilogue: divide by softmax denominator, scatter O^T back row-major.
#pragma unroll
  for (int qt = 0; qt < 2; ++qt) {
    const float inv = 1.0f / lq[qt];
    float* orow = out + ((size_t)(b * TT + m0 + 16 * qt + ql)) * DHD;
#pragma unroll
    for (int t = 0; t < 4; ++t)
#pragma unroll
      for (int r = 0; r < 8; ++r)
        orow[16 * t + r + 8 * h] = o[qt][t][r] * inv;
  }
}

// ---------------------------------------------------------------------------
extern "C" void kernel_launch(void* const* d_in, const int* in_sizes, int n_in,
                              void* d_out, int out_size, void* d_ws, size_t ws_size,
                              hipStream_t stream)
{
  const float* x    = (const float*)d_in[0];
  const float* Wq   = (const float*)d_in[1];
  const float* Wk   = (const float*)d_in[2];
  const float* Wv   = (const float*)d_in[3];
  const float* Wq_s = (const float*)d_in[4];
  const float* Wk_s = (const float*)d_in[5];
  const float* Wv_s = (const float*)d_in[6];
  float* out = (float*)d_out;

  const size_t nElem = (size_t)BATCH * TT * DHD;           // 1,048,576
  __bf16* Qb = (__bf16*)d_ws;                              // 2 MB
  __bf16* Kb = (__bf16*)((char*)d_ws + 2 * nElem);         // 2 MB
  __bf16* Vt = (__bf16*)((char*)d_ws + 4 * nElem);         // 2 MB (transposed [B][64][T])

  proj_kernel<<<dim3(BATCH * (TT / 16)), dim3(128), 0, stream>>>(x, Wq, Wk, Wv, Qb, Kb, Vt);
  fixup_kernel<<<dim3(BATCH * 16), dim3(64), 0, stream>>>(x, Wq_s, Wk_s, Wv_s, Qb, Kb, Vt);
  attn_kernel<<<dim3(TT / 128, BATCH), dim3(128), 0, stream>>>(Qb, Kb, Vt, out);
}
